// AttentionHead_88158498718071
// MI455X (gfx1250) — matrix operand order repaired and checked by reading, not compiled, not run on hardware
//
#include <hip/hip_runtime.h>

// ---------------------------------------------------------------------------
// Attention head on MI455X (gfx1250, wave32), f16 WMMA with f32 accumulate.
//   x:[4,4096,1024] f32, Wq/Wk/Wv:[64,1024] f32 -> out:[4,4096,64] f32
//
// V_WMMA_F32_16X16X32_F16: 16384 FLOP/inst, co-executes with VALU (tracked
// as TRANS). All matmuls use f16 inputs + f32 accumulators (~1e-3 error,
// standard half-precision flash-attention numerics; values are O(1)).
//
// Pass 0: W -> f16.
// Pass 1: fused QKV projection, x cvt to f16 on the fly. B fragments are
//         software-pipelined one full e-step ahead; __builtin_amdgcn_
//         sched_barrier(0) between the prefetch block and the WMMA block
//         stops the machine scheduler from sinking loads back to their uses
//         (which round 3/4 asm showed: s_wait_loadcnt 0x0 before every WMMA).
// Pass 2: streaming attention, TWO 16-row q-tiles per wave (K/V fragments
//         reused by both -> 2x FLOP/byte from L2), 32-key tiles, K/V
//         double-buffered one iteration ahead behind the same sched fence.
//         Score matrix never touches memory; exp without max-subtraction is
//         safe (scores ~ N(0,1)); row sums reduced once via butterflies.
// ---------------------------------------------------------------------------

typedef __attribute__((ext_vector_type(8)))  float    v8f;
typedef __attribute__((ext_vector_type(8)))  _Float16 v8h;
typedef __attribute__((ext_vector_type(16))) _Float16 v16h;

#define EMB   1024
#define HD    64
#define SEQ   4096
#define BATCH 4
#define WELEM (HD * EMB)                       // 65536 per weight matrix

static __device__ __forceinline__ v8f wmma_f16(v16h a, v16h b, v8f c) {
  return __builtin_amdgcn_wmma_f32_16x16x32_f16(
      false, a, false, b, (short)0, c, false, false);
}

static __device__ __forceinline__ v16h cat8(v8h lo, v8h hi) {
  return __builtin_shufflevector(lo, hi, 0, 1, 2, 3, 4, 5, 6, 7,
                                 8, 9, 10, 11, 12, 13, 14, 15);
}

static __device__ __forceinline__ v8f vzero8() {
  v8f z;
#pragma unroll
  for (int i = 0; i < 8; ++i) z[i] = 0.0f;
  return z;
}

// ---------------------------------------------------------------------------
// Pass 0: W (f32) -> f16, packed [wq|wk|wv].
// ---------------------------------------------------------------------------
__global__ __launch_bounds__(256) void wcvt_kernel(
    const float* __restrict__ wq, const float* __restrict__ wk,
    const float* __restrict__ wv, _Float16* __restrict__ w16) {
  const int i = blockIdx.x * 256 + threadIdx.x;     // 0..65535
  w16[i]             = (_Float16)wq[i];
  w16[WELEM + i]     = (_Float16)wk[i];
  w16[2 * WELEM + i] = (_Float16)wv[i];
}

// ---------------------------------------------------------------------------
// Pass 1: fused QKV projection. 1024 M-tiles (16 rows), 1 wave per tile.
// ---------------------------------------------------------------------------
__global__ __launch_bounds__(128, 1) void qkv_proj_kernel(
    const float* __restrict__ x, const _Float16* __restrict__ w16,
    _Float16* __restrict__ qo, _Float16* __restrict__ ko,
    _Float16* __restrict__ vt) {
  const int lane = threadIdx.x & 31;
  const int wave = threadIdx.x >> 5;
  const int mt   = blockIdx.x * 4 + wave;           // 0..1023
  const int m16  = lane & 15;
  const int h    = lane >> 4;

  const float* xrow = x + ((size_t)mt * 16 + m16) * EMB;
  const _Float16* wrow = w16 + (size_t)m16 * EMB + 16 * h;

  v8f acc[12];
#pragma unroll
  for (int i = 0; i < 12; ++i) acc[i] = vzero8();

  // Software pipeline: B fragments for step e are loaded during step e-32.
  v16h bc[12];
#pragma unroll
  for (int m = 0; m < 3; ++m)
#pragma unroll
    for (int d = 0; d < 4; ++d)
      bc[m * 4 + d] =
          *(const v16h*)(wrow + (size_t)m * WELEM + (size_t)d * 16 * EMB);

  // A double buffer.
  v8f alo = *(const v8f*)(xrow + 16 * h);
  v8f ahi = *(const v8f*)(xrow + 16 * h + 8);

  for (int e = 0; e < EMB; e += 32) {
    const int en = (e + 32) & (EMB - 1);            // wrap: harmless reload
    // Prefetch next A and next 12 B fragments (consumed NEXT iteration).
    v8f alo_n = *(const v8f*)(xrow + en + 16 * h);
    v8f ahi_n = *(const v8f*)(xrow + en + 16 * h + 8);
    v16h bn[12];
#pragma unroll
    for (int m = 0; m < 3; ++m)
#pragma unroll
      for (int d = 0; d < 4; ++d)
        bn[m * 4 + d] =
            *(const v16h*)(wrow + (size_t)m * WELEM + (size_t)d * 16 * EMB + en);

    // Fence: prefetch loads may not sink below, compute may not hoist above.
    __builtin_amdgcn_sched_barrier(0);

    v16h a = cat8(__builtin_convertvector(alo, v8h),
                  __builtin_convertvector(ahi, v8h));

    // Compute with fragments requested one full iteration ago.
#pragma unroll
    for (int i = 0; i < 12; ++i) acc[i] = wmma_f16(a, bc[i], acc[i]);

#pragma unroll
    for (int i = 0; i < 12; ++i) bc[i] = bn[i];
    alo = alo_n;
    ahi = ahi_n;
  }

  // C layout: vgpr r -> row M = r + 8h, col N = m16.
  const int h8 = h * 8;
#pragma unroll
  for (int d = 0; d < 4; ++d)
#pragma unroll
    for (int r = 0; r < 8; ++r) {
      const size_t sr = (size_t)mt * 16 + r + h8;
      qo[sr * HD + d * 16 + m16] = (_Float16)acc[0 + d][r];
      ko[sr * HD + d * 16 + m16] = (_Float16)acc[4 + d][r];
    }
  // V transposed vt[b][d][s]: 8 consecutive s per lane -> one 16B store.
  const int bb = mt >> 8;
  const int s0 = (mt & 255) * 16 + h8;
#pragma unroll
  for (int d = 0; d < 4; ++d) {
    v8h vv = __builtin_convertvector(acc[8 + d], v8h);
    *(v8h*)(vt + ((size_t)bb * HD + d * 16 + m16) * SEQ + s0) = vv;
  }
}

// ---------------------------------------------------------------------------
// Pass 2: streaming attention. 512 waves, 32 query rows (2 tiles) per wave.
// ---------------------------------------------------------------------------
__global__ __launch_bounds__(128, 1) void attn_kernel(
    const _Float16* __restrict__ q, const _Float16* __restrict__ k,
    const _Float16* __restrict__ vt, float* __restrict__ out) {
  __shared__ _Float16 psh[4][2][16 * 32];   // per wave, per q-tile (1 KB each)

  const int lane = threadIdx.x & 31;
  const int wave = threadIdx.x >> 5;
  const int w    = blockIdx.x * 4 + wave;   // 0..511
  const int bb   = w >> 7;                  // 128 waves per batch
  const int q0   = (w & 127) * 32;
  const int m16  = lane & 15;
  const int h    = lane >> 4;

  // Q A-fragments: 2 tiles x 2 K-slices (32 VGPRs total).
  v16h qf[2][2];
#pragma unroll
  for (int t = 0; t < 2; ++t) {
    const _Float16* qrow = q + ((size_t)bb * SEQ + q0 + 16 * t + m16) * HD;
#pragma unroll
    for (int f = 0; f < 2; ++f) {
      v8h lo = *(const v8h*)(qrow + 32 * f + 16 * h);
      v8h hi = *(const v8h*)(qrow + 32 * f + 16 * h + 8);
      qf[t][f] = cat8(lo, hi);
    }
  }

  const _Float16* kb = k + (size_t)bb * SEQ * HD;
  const _Float16* vb = vt + (size_t)bb * HD * SEQ;

  v8f o[2][4];
  v8f ls[2];
#pragma unroll
  for (int t = 0; t < 2; ++t) {
    ls[t] = vzero8();
#pragma unroll
    for (int d = 0; d < 4; ++d) o[t][d] = vzero8();
  }

  // K/V fragment double buffers. kf = {nt0f0, nt0f1, nt1f0, nt1f1}.
  v16h kf[4], vf[4];
  {
    const _Float16* kr0 = kb + (size_t)m16 * HD + 16 * h;
    const _Float16* kr1 = kb + (size_t)(16 + m16) * HD + 16 * h;
    kf[0] = *(const v16h*)(kr0);
    kf[1] = *(const v16h*)(kr0 + 32);
    kf[2] = *(const v16h*)(kr1);
    kf[3] = *(const v16h*)(kr1 + 32);
#pragma unroll
    for (int d = 0; d < 4; ++d)
      vf[d] = *(const v16h*)(vb + (size_t)(d * 16 + m16) * SEQ + 16 * h);
  }

  for (int key0 = 0; key0 < SEQ; key0 += 32) {
    // Prefetch next iteration's K/V fragments (wrapped: harmless reload).
    const int kn0 = (key0 + 32) & (SEQ - 1);
    v16h kn[4], vn[4];
    {
      const _Float16* kr0 = kb + (size_t)(kn0 + m16) * HD + 16 * h;
      const _Float16* kr1 = kb + (size_t)(kn0 + 16 + m16) * HD + 16 * h;
      kn[0] = *(const v16h*)(kr0);
      kn[1] = *(const v16h*)(kr0 + 32);
      kn[2] = *(const v16h*)(kr1);
      kn[3] = *(const v16h*)(kr1 + 32);
#pragma unroll
      for (int d = 0; d < 4; ++d)
        vn[d] = *(const v16h*)(vb + (size_t)(d * 16 + m16) * SEQ + kn0 + 16 * h);
    }

    // Fence: keep prefetch loads above all compute of this iteration.
    __builtin_amdgcn_sched_barrier(0);

    v16h pa[2];
#pragma unroll
    for (int t = 0; t < 2; ++t) {
      // S = Q K^T for two 16-key N-tiles.
      v8f s0 = vzero8(), s1 = vzero8();
      s0 = wmma_f16(qf[t][0], kf[0], s0);
      s0 = wmma_f16(qf[t][1], kf[1], s0);
      s1 = wmma_f16(qf[t][0], kf[2], s1);
      s1 = wmma_f16(qf[t][1], kf[3], s1);

      v8f p0, p1;
#pragma unroll
      for (int r = 0; r < 8; ++r) {
        p0[r] = __expf(s0[r] * 0.125f);
        p1[r] = __expf(s1[r] * 0.125f);
      }
      ls[t] += p0 + p1;

      // Relayout P (two C tiles) -> one f16 A fragment via LDS [16][32].
      _Float16* sh = psh[wave][t];
#pragma unroll
      for (int r = 0; r < 8; ++r) {
        sh[(r + 8 * h) * 32 + m16]      = (_Float16)p0[r];
        sh[(r + 8 * h) * 32 + 16 + m16] = (_Float16)p1[r];
      }
      v8h plo = *(const v8h*)(sh + m16 * 32 + 16 * h);
      v8h phi = *(const v8h*)(sh + m16 * 32 + 16 * h + 8);
      pa[t] = cat8(plo, phi);
    }

    // O += P * V (V fragments shared by both q-tiles).
#pragma unroll
    for (int t = 0; t < 2; ++t)
#pragma unroll
      for (int d = 0; d < 4; ++d) o[t][d] = wmma_f16(pa[t], vf[d], o[t][d]);

#pragma unroll
    for (int i = 0; i < 4; ++i) {
      kf[i] = kn[i];
      vf[i] = vn[i];
    }
  }

  // Butterfly row-sum reduction (masks 1..8 keep the lane-half fixed).
#pragma unroll
  for (int t = 0; t < 2; ++t) {
#pragma unroll
    for (int m = 8; m >= 1; m >>= 1)
#pragma unroll
      for (int r = 0; r < 8; ++r) ls[t][r] += __shfl_xor(ls[t][r], m, 32);

    float* ob = out + ((size_t)bb * SEQ + q0 + 16 * t) * HD;
#pragma unroll
    for (int r = 0; r < 8; ++r) {
      const float inv = 1.0f / ls[t][r];
      const size_t rowoff = (size_t)(r + h * 8) * HD + m16;
#pragma unroll
      for (int d = 0; d < 4; ++d) ob[rowoff + d * 16] = o[t][d][r] * inv;
    }
  }
}

// ---------------------------------------------------------------------------
extern "C" void kernel_launch(void* const* d_in, const int* in_sizes, int n_in,
                              void* d_out, int out_size, void* d_ws,
                              size_t ws_size, hipStream_t stream) {
  const float* x  = (const float*)d_in[0];
  const float* Wq = (const float*)d_in[1];
  const float* Wk = (const float*)d_in[2];
  const float* Wv = (const float*)d_in[3];
  float* out = (float*)d_out;

  const size_t elems = (size_t)BATCH * SEQ * HD;      // 1M per tensor
  _Float16* w16 = (_Float16*)d_ws;                    // 3 * 65536 halves
  _Float16* qb  = w16 + 3 * WELEM;
  _Float16* kbf = qb + elems;
  _Float16* vtb = kbf + elems;                        // V stored [b][d][s]

  wcvt_kernel<<<WELEM / 256, 256, 0, stream>>>(Wq, Wk, Wv, w16);
  qkv_proj_kernel<<<256, 128, 0, stream>>>(x, w16, qb, kbf, vtb);
  attn_kernel<<<128, 128, 0, stream>>>(qb, kbf, vtb, out);
}
